// CharPredictorMultirateFFN_72928544686167
// MI455X (gfx1250) — compile-verified, matrix operand order checked by reading, not run on hardware
//
#include <hip/hip_runtime.h>
#include <hip/hip_bf16.h>

#define BDIM 32
#define SDIM 2048
#define VDIM 256
#define EDIM 512
#define HDIM 1024
#define KTAP 16
#define SPAD (SDIM + 16)      // 15 leading zero rows + 1 spare
#define FDIM (EDIM + HDIM)    // 1536

typedef __attribute__((ext_vector_type(16))) __bf16 v16bf;
typedef __attribute__((ext_vector_type(8)))  float  v8f;
typedef __attribute__((ext_vector_type(4)))  unsigned int v4u;
typedef __attribute__((ext_vector_type(8)))  int v8i;
typedef __attribute__((ext_vector_type(4)))  int v4i;

union Frag { v16bf v; uint4 q[2]; };

__device__ __forceinline__ unsigned short f2bf(float f) {
  union { float f; unsigned u; } x; x.f = f;
  unsigned r = x.u + 0x7FFFu + ((x.u >> 16) & 1u);   // round-to-nearest-even
  return (unsigned short)(r >> 16);
}

// Issue a TDM 2-D tile load (bf16 elements) global -> LDS.
// dim0 = row length (elements), rows = #rows, stride = row stride (elements).
__device__ __forceinline__ void tdm_load_2d(unsigned lds_addr, const void* gptr,
                                            unsigned dim0, unsigned rows,
                                            unsigned stride) {
  const unsigned long long ga = (unsigned long long)gptr;
  v4u g0;
  g0.x = 1u;                                             // count=1, user mode
  g0.y = lds_addr;                                       // LDS byte address
  g0.z = (unsigned)(ga & 0xFFFFFFFFu);                   // global_addr[31:0]
  g0.w = (unsigned)((ga >> 32) & 0x01FFFFFFu) | (2u << 30); // addr[56:32] | type=2
  v8i g1;
  g1[0] = (int)(1u << 16);                               // data_size=1 (2 bytes)
  g1[1] = (int)((dim0 & 0xFFFFu) << 16);                 // tensor_dim0[15:0] @ bit48
  g1[2] = (int)(((dim0 >> 16) & 0xFFFFu) | ((rows & 0xFFFFu) << 16)); // dim0 hi | dim1 lo
  g1[3] = (int)(((rows >> 16) & 0xFFFFu) | ((dim0 & 0xFFFFu) << 16)); // dim1 hi | tile_dim0
  g1[4] = (int)(rows & 0xFFFFu);                         // tile_dim1 | tile_dim2=0
  g1[5] = (int)stride;                                   // tensor_dim0_stride[31:0]
  g1[6] = 0;                                             // stride hi | dim1_stride lo
  g1[7] = 0;
  v4i z4 = {0, 0, 0, 0};
  v8i z8 = {0, 0, 0, 0, 0, 0, 0, 0};
  __builtin_amdgcn_tensor_load_to_lds(g0, g1, z4, z4, z8, 0);
}

// ---------------- prep kernels ----------------

__global__ void embed_pad_kernel(const int* __restrict__ seq,
                                 const float* __restrict__ table,
                                 unsigned short* __restrict__ embp) {
  const int row = blockIdx.x;            // 0 .. B*SPAD-1
  const int b = row / SPAD, p = row % SPAD;
  const int tid = threadIdx.x;           // 256 threads, 2 elems each
  const size_t base = (size_t)row * EDIM;
  if (p < 15 || p >= SDIM + 15) {
    embp[base + tid] = 0;
    embp[base + tid + 256] = 0;
  } else {
    const int s = p - 15;
    const int v = seq[(size_t)b * SDIM + s];
    const float* src = table + (size_t)v * EDIM;
    embp[base + tid]       = f2bf(src[tid]);
    embp[base + tid + 256] = f2bf(src[tid + 256]);
  }
}

// wct[k][h][e] = bf16(conv_w[h][e][k])  -> GEMM-K (e) contiguous per h row
__global__ void pack_convw_kernel(const float* __restrict__ w,
                                  unsigned short* __restrict__ wct) {
  const size_t o = (size_t)blockIdx.x * blockDim.x + threadIdx.x;
  if (o >= (size_t)KTAP * HDIM * EDIM) return;
  const int e = (int)(o % EDIM);
  const int h = (int)((o / EDIM) % HDIM);
  const int k = (int)(o / ((size_t)EDIM * HDIM));
  wct[o] = f2bf(w[((size_t)h * EDIM + e) * KTAP + k]);
}

__global__ void pack_linw_kernel(const float* __restrict__ w,
                                 unsigned short* __restrict__ lw) {
  const size_t o = (size_t)blockIdx.x * blockDim.x + threadIdx.x;
  if (o < (size_t)VDIM * FDIM) lw[o] = f2bf(w[o]);
}

// ---------------- conv as 16 tap-GEMMs (WMMA bf16) ----------------

__global__ void __launch_bounds__(256)
conv_wmma_kernel(const unsigned short* __restrict__ embp,
                 const unsigned short* __restrict__ wct,
                 const float* __restrict__ conv_b,
                 unsigned short* __restrict__ convo) {
  __shared__ unsigned short sEmb[31 * EDIM];          // 31 KB: rows t0..t0+30 (padded coords)

  const int blk = blockIdx.x;                         // B * S/16 blocks
  const int b  = blk / (SDIM / 16);
  const int t0 = (blk % (SDIM / 16)) * 16;
  const int tid = threadIdx.x;
  const int wave  = tid >> 5;
  const int lane  = tid & 31;
  const int half  = lane >> 4;        // K-half select per ISA fragment layout
  const int row16 = lane & 15;        // M (for A) / N (for B)

  // TDM: DMA the 31x512 bf16 embedding tile into LDS (wave 0 issues + waits).
  if (wave == 0) {
    tdm_load_2d((unsigned)(size_t)&sEmb[0],
                embp + ((size_t)b * SPAD + t0) * EDIM,
                EDIM, 31, EDIM);
    __builtin_amdgcn_s_wait_tensorcnt(0);
  }
  __syncthreads();

  // 8 accumulators: whole 128-channel strip per wave, A fragment loaded once.
  v8f acc[8];
  #pragma unroll
  for (int i = 0; i < 8; ++i) acc[i] = (v8f){};
  const int hbase = wave * 128 + row16;

  for (int k = 0; k < KTAP; ++k) {
    const unsigned short* arow  = sEmb + (size_t)(row16 + k) * EDIM;
    const unsigned short* brow0 = wct + ((size_t)k * HDIM + hbase) * EDIM;
    for (int e0 = 0; e0 < EDIM; e0 += 32) {
      Frag a;
      a.q[0] = *(const uint4*)(arow + e0 + half * 8);
      a.q[1] = *(const uint4*)(arow + e0 + 16 + half * 8);
      #pragma unroll
      for (int nt = 0; nt < 8; ++nt) {
        Frag bm;
        const unsigned short* brow = brow0 + (size_t)nt * 16 * EDIM;
        bm.q[0] = *(const uint4*)(brow + e0 + half * 8);
        bm.q[1] = *(const uint4*)(brow + e0 + 16 + half * 8);
        acc[nt] = __builtin_amdgcn_wmma_f32_16x16x32_bf16(false, a.v, false, bm.v,
                                                          (short)0, acc[nt], false, false);
      }
    }
  }

  #pragma unroll
  for (int nt = 0; nt < 8; ++nt) {
    const int h = wave * 128 + nt * 16 + row16;
    const float bias = conv_b[h];
    #pragma unroll
    for (int r = 0; r < 8; ++r) {     // C layout: M = r + half*8, N = row16
      const int t = t0 + r + half * 8;
      float y = acc[nt][r] + bias;
      y = y > 0.f ? y : 0.f;          // relu
      convo[((size_t)b * SDIM + t) * HDIM + h] = f2bf(y);
    }
  }
}

// ---------------- linear (WMMA) + softmax, fused ----------------

__global__ void __launch_bounds__(256)
linear_softmax_kernel(const unsigned short* __restrict__ embp,
                      const unsigned short* __restrict__ convo,
                      const unsigned short* __restrict__ lw,
                      const float* __restrict__ lin_b,
                      float* __restrict__ out) {
  __shared__ unsigned short sFeatE[16 * EDIM];  // 16 KB: embedding part
  __shared__ unsigned short sFeatH[16 * HDIM];  // 32 KB: conv part
  __shared__ float sLog[16 * VDIM];             // 16 KB logits

  const int blk = blockIdx.x;
  const int b  = blk / (SDIM / 16);
  const int t0 = (blk % (SDIM / 16)) * 16;
  const int tid = threadIdx.x;
  const int wave  = tid >> 5;
  const int lane  = tid & 31;
  const int half  = lane >> 4;
  const int row16 = lane & 15;

  // TDM: two rectangular tiles -> two LDS regions (wave 0 issues + waits).
  if (wave == 0) {
    tdm_load_2d((unsigned)(size_t)&sFeatE[0],
                embp + ((size_t)b * SPAD + t0 + 15) * EDIM,
                EDIM, 16, EDIM);
    tdm_load_2d((unsigned)(size_t)&sFeatH[0],
                convo + ((size_t)b * SDIM + t0) * HDIM,
                HDIM, 16, HDIM);
    __builtin_amdgcn_s_wait_tensorcnt(0);
  }
  __syncthreads();

  v8f acc[2];
  acc[0] = (v8f){};
  acc[1] = (v8f){};
  const int vbase = wave * 32 + row16;          // wave covers 2 V-tiles (V=256)

  // Region 1: embedding features (f = 0..511)
  for (int e0 = 0; e0 < EDIM; e0 += 32) {
    Frag a;
    const unsigned short* arow = sFeatE + (size_t)row16 * EDIM;
    a.q[0] = *(const uint4*)(arow + e0 + half * 8);
    a.q[1] = *(const uint4*)(arow + e0 + 16 + half * 8);
    #pragma unroll
    for (int nt = 0; nt < 2; ++nt) {
      Frag bm;
      const unsigned short* brow = lw + (size_t)(vbase + nt * 16) * FDIM + e0;
      bm.q[0] = *(const uint4*)(brow + half * 8);
      bm.q[1] = *(const uint4*)(brow + 16 + half * 8);
      acc[nt] = __builtin_amdgcn_wmma_f32_16x16x32_bf16(false, a.v, false, bm.v,
                                                        (short)0, acc[nt], false, false);
    }
  }
  // Region 2: conv features (f = 512..1535)
  for (int e0 = 0; e0 < HDIM; e0 += 32) {
    Frag a;
    const unsigned short* arow = sFeatH + (size_t)row16 * HDIM;
    a.q[0] = *(const uint4*)(arow + e0 + half * 8);
    a.q[1] = *(const uint4*)(arow + e0 + 16 + half * 8);
    #pragma unroll
    for (int nt = 0; nt < 2; ++nt) {
      Frag bm;
      const unsigned short* brow = lw + (size_t)(vbase + nt * 16) * FDIM + EDIM + e0;
      bm.q[0] = *(const uint4*)(brow + half * 8);
      bm.q[1] = *(const uint4*)(brow + 16 + half * 8);
      acc[nt] = __builtin_amdgcn_wmma_f32_16x16x32_bf16(false, a.v, false, bm.v,
                                                        (short)0, acc[nt], false, false);
    }
  }

  #pragma unroll
  for (int nt = 0; nt < 2; ++nt) {
    const int v = vbase + nt * 16;
    const float bias = lin_b[v];
    #pragma unroll
    for (int r = 0; r < 8; ++r)
      sLog[(r + half * 8) * VDIM + v] = acc[nt][r] + bias;
  }
  __syncthreads();

  // Row-wise softmax over V=256. Thread tid handles row tid/16, 16 contiguous cols.
  // Threads sharing a row form a 16-lane group inside one wave32 -> shfl_xor reduce.
  const int r  = tid >> 4;
  const int cg = tid & 15;
  const float* lrow = sLog + r * VDIM + cg * 16;
  float vals[16];
  float mx = -1e30f;
  #pragma unroll
  for (int j = 0; j < 16; ++j) { vals[j] = lrow[j]; mx = fmaxf(mx, vals[j]); }
  #pragma unroll
  for (int off = 1; off < 16; off <<= 1) mx = fmaxf(mx, __shfl_xor(mx, off));
  float sum = 0.f;
  #pragma unroll
  for (int j = 0; j < 16; ++j) { vals[j] = __expf(vals[j] - mx); sum += vals[j]; }
  #pragma unroll
  for (int off = 1; off < 16; off <<= 1) sum += __shfl_xor(sum, off);
  const float inv = 1.0f / sum;
  float* orow = out + ((size_t)b * SDIM + t0 + r) * VDIM + cg * 16;
  #pragma unroll
  for (int j = 0; j < 16; ++j) orow[j] = vals[j] * inv;
}

// ---------------- launcher ----------------

extern "C" void kernel_launch(void* const* d_in, const int* in_sizes, int n_in,
                              void* d_out, int out_size, void* d_ws, size_t ws_size,
                              hipStream_t stream) {
  (void)in_sizes; (void)n_in; (void)out_size; (void)ws_size;
  const int*   seq   = (const int*)  d_in[0];
  const float* table = (const float*)d_in[1];
  const float* convw = (const float*)d_in[2];
  const float* convb = (const float*)d_in[3];
  const float* linw  = (const float*)d_in[4];
  const float* linb  = (const float*)d_in[5];
  float* out = (float*)d_out;

  char* ws = (char*)d_ws;
  size_t off = 0;
  auto carve = [&](size_t bytes) -> void* {
    size_t a = (off + 255) & ~(size_t)255;
    off = a + bytes;
    return ws + a;
  };
  unsigned short* embp = (unsigned short*)carve((size_t)BDIM * SPAD * EDIM * 2); // ~67.6 MB
  unsigned short* wct  = (unsigned short*)carve((size_t)KTAP * HDIM * EDIM * 2); // 16 MB
  unsigned short* lwbf = (unsigned short*)carve((size_t)VDIM * FDIM * 2);        // 0.75 MB
  unsigned short* convo= (unsigned short*)carve((size_t)BDIM * SDIM * HDIM * 2); // 128 MB

  embed_pad_kernel<<<BDIM * SPAD, 256, 0, stream>>>(seq, table, embp);
  pack_convw_kernel<<<(KTAP * HDIM * EDIM + 255) / 256, 256, 0, stream>>>(convw, wct);
  pack_linw_kernel<<<(VDIM * FDIM + 255) / 256, 256, 0, stream>>>(linw, lwbf);

  const int tiles = BDIM * (SDIM / 16);   // 4096
  conv_wmma_kernel<<<tiles, 256, 0, stream>>>(embp, wct, convb, convo);
  linear_softmax_kernel<<<tiles, 256, 0, stream>>>(embp, convo, lwbf, linb, out);
}
